// MessageFunction_50543175139483
// MI455X (gfx1250) — compile-verified, matrix operand order checked by reading, not run on hardware
//
#include <hip/hip_runtime.h>
#include <hip/hip_bf16.h>

// ---------------------------------------------------------------------------
// Fused per-edge MLP message function for MI455X (gfx1250, wave32, WMMA).
//
//   m[e,i] = sum_j (selu(e*W_A1+b_A1) @ W_A2 + b_A2)[e, i*32+j] * h[e,j]
//          + (selu(e*W_b1+b_b1) @ W_b2 + b_b2)[e,i]
//
// One wave processes 32 edges as TWO 16-edge groups: each W_A2^T fragment is
// loaded once and feeds two back-to-back WMMAs (2x L2 reuse). Fragment loads
// are double-buffered (tile t+1 in flight across tile t's 8 WMMAs).
// ---------------------------------------------------------------------------

typedef __attribute__((ext_vector_type(16))) _Float16 v16h;
typedef __attribute__((ext_vector_type(8)))  float    v8f;

union F16x16 { v16h v; float4 f4[2]; };

// 16-bit A/B fragment K-slot mapping (ISA 7.12.2, 16-bit A 16x32):
//   element q (0..15), lane-half hi: K = (q<8 ? q : q+8) + hi*8
__device__ __host__ __forceinline__ int klocal(int q, int hi) {
    return (q < 8 ? q : q + 8) + hi * 8;
}

__device__ __forceinline__ float selu_f(float x) {
    const float alpha = 1.6732632423543772f;
    const float scale = 1.0507009873554805f;
    float neg = alpha * (__expf(x) - 1.0f);
    return scale * (x > 0.0f ? x : neg);
}

// ---------------- workspace layout (bytes) ----------------
#define OFF_WT    0         // 256 frags * 1024B  : W_A2^T fragments (f16)
#define OFF_WB2   262144    //   8 frags * 1024B  : W_b2^T fragments (f16)
#define OFF_BA2   270336    //   2 frags * 1024B  : b_A2  fragments (f16)
#define OFF_W1A   272384    // 4*32*16 f32        : W_A1 in fragment order
#define OFF_B1A   280576    // 4*32*16 f32        : b_A1
#define OFF_W1B   288768    // 4*32*16 f32        : W_b1
#define OFF_B1B   296960    // 4*32*16 f32        : b_b1

// Pack W_A2 [128,1024] into lane-major f16 A-fragments for D = W^T x SA^T.
__global__ void pack_wtA(const float* __restrict__ W_A2,
                         _Float16* __restrict__ wt) {
    int tid  = blockIdx.x * blockDim.x + threadIdx.x;   // 131072 threads
    int frag = tid >> 9;
    int lane = (tid >> 4) & 31;
    int q    = tid & 15;
    int hi = lane >> 4, m = lane & 15;
    int c = frag & 3, t = frag >> 2;
    int k = c * 32 + klocal(q, hi);
    int n = t * 16 + m;
    wt[frag * 512 + lane * 16 + q] = (_Float16)W_A2[k * 1024 + n];
}

__global__ void pack_aux(const float* __restrict__ W_b2,
                         const float* __restrict__ b_A2,
                         const float* __restrict__ W_A1,
                         const float* __restrict__ b_A1,
                         const float* __restrict__ W_b1,
                         const float* __restrict__ b_b1,
                         _Float16* __restrict__ wb2p,
                         _Float16* __restrict__ ba2p,
                         float* __restrict__ w1Ap, float* __restrict__ b1Ap,
                         float* __restrict__ w1Bp, float* __restrict__ b1Bp) {
    int tid = blockIdx.x * blockDim.x + threadIdx.x;    // 7168 threads
    if (tid < 4096) {                                   // W_b2^T: 8 frags
        int frag = tid >> 9, lane = (tid >> 4) & 31, q = tid & 15;
        int hi = lane >> 4, m = lane & 15;
        int c = frag & 3, T = frag >> 2;
        int k = c * 32 + klocal(q, hi);
        int i = T * 16 + m;
        wb2p[frag * 512 + lane * 16 + q] = (_Float16)W_b2[k * 32 + i];
    } else if (tid < 5120) {                            // b_A2: 2 frags (K=32)
        int u = tid - 4096;
        int frag = u >> 9, lane = (u >> 4) & 31, q = u & 15;
        int hi = lane >> 4, m = lane & 15;
        int j = klocal(q, hi);
        int i = frag * 16 + m;
        ba2p[frag * 512 + lane * 16 + q] = (_Float16)b_A2[i * 32 + j];
    } else {                                            // lin1 weights: 2048
        int u = tid - 5120;                             // u = (c*32+lane)*16+q
        int lane = (u >> 4) & 31, q = u & 15;
        int hi = lane >> 4;
        int c = u >> 9;
        int k = c * 32 + klocal(q, hi);
        w1Ap[u] = W_A1[k];  b1Ap[u] = b_A1[k];
        w1Bp[u] = W_b1[k];  b1Bp[u] = b_b1[k];
    }
}

__device__ __forceinline__ void load_frag(F16x16& wf, const _Float16* base,
                                          int frag, int lane) {
    const float4* fp = (const float4*)(base + frag * 512 + lane * 16);
    wf.f4[0] = fp[0];
    wf.f4[1] = fp[1];
}

__global__ __launch_bounds__(256) void edge_mlp_wmma(
    const float* __restrict__ h_i, const float* __restrict__ e_ij,
    const _Float16* __restrict__ wtA, const _Float16* __restrict__ wb2p,
    const _Float16* __restrict__ ba2p,
    const float* __restrict__ w1Ap, const float* __restrict__ b1Ap,
    const float* __restrict__ w1Bp, const float* __restrict__ b1Bp,
    const float* __restrict__ b_b2, float* __restrict__ out) {
    const int lane = threadIdx.x & 31;
    const int wave = threadIdx.x >> 5;
    const int blk  = blockIdx.x * (blockDim.x >> 5) + wave;   // 32-edge block
    const int m  = lane & 15;
    const int hi = lane >> 4;
    const long e0 = (long)blk * 32 + m;       // group-0 edge for this lane
    const long e1 = e0 + 16;                  // group-1 edge for this lane

    // Per-lane h subsets: hreg[q] = h[e, klocal(q,hi)] -- exactly the 16 j's
    // this lane multiplies (and simultaneously the h B-fragment slots).
    float hreg0[16], hreg1[16];
#pragma unroll
    for (int g = 0; g < 2; ++g) {
        float* hr = g ? hreg1 : hreg0;
        const float* hp = h_i + (g ? e1 : e0) * 32;
        float4 a = *(const float4*)(hp + hi * 8);
        float4 b = *(const float4*)(hp + hi * 8 + 4);
        float4 c = *(const float4*)(hp + 16 + hi * 8);
        float4 d = *(const float4*)(hp + 16 + hi * 8 + 4);
        hr[0]=a.x; hr[1]=a.y; hr[2]=a.z;  hr[3]=a.w;
        hr[4]=b.x; hr[5]=b.y; hr[6]=b.z;  hr[7]=b.w;
        hr[8]=c.x; hr[9]=c.y; hr[10]=c.z; hr[11]=c.w;
        hr[12]=d.x;hr[13]=d.y;hr[14]=d.z; hr[15]=d.w;
    }

    const float s0 = e_ij[e0];
    const float s1 = e_ij[e1];

    // In-register selu MLP -> B-operand fragments SA^T for both groups.
    F16x16 fa0[4], fa1[4];
#pragma unroll
    for (int c = 0; c < 4; ++c) {
        const float* wp = w1Ap + (c * 32 + lane) * 16;
        const float* bp = b1Ap + (c * 32 + lane) * 16;
#pragma unroll
        for (int q = 0; q < 16; ++q) {
            float w = wp[q], b = bp[q];
            fa0[c].v[q] = (_Float16)selu_f(fmaf(s0, w, b));
            fa1[c].v[q] = (_Float16)selu_f(fmaf(s1, w, b));
        }
    }

    float P0[32], P1[32];
#pragma unroll
    for (int i = 0; i < 32; ++i) { P0[i] = 0.0f; P1[i] = 0.0f; }

    // ---- main term: 64 n-tiles x 4 K-chunks, 2 WMMAs per fragment ----
    // Double-buffered fragment loads: tile t+1 is in flight across the
    // 8 WMMAs + 16 VALU FMAs of tile t.
    F16x16 wbuf[2][4];
#pragma unroll
    for (int c = 0; c < 4; ++c) load_frag(wbuf[0][c], wtA, c, lane);

#pragma unroll
    for (int t = 0; t < 64; ++t) {
        const int cur = t & 1, nxt = cur ^ 1;
        if (t < 63) {
#pragma unroll
            for (int c = 0; c < 4; ++c)
                load_frag(wbuf[nxt][c], wtA, (t + 1) * 4 + c, lane);
        }
        v8f acc0 = {}, acc1 = {};
#pragma unroll
        for (int c = 0; c < 4; ++c) {
            acc0 = __builtin_amdgcn_wmma_f32_16x16x32_f16(
                false, wbuf[cur][c].v, false, fa0[c].v, (short)0, acc0,
                false, false);
            acc1 = __builtin_amdgcn_wmma_f32_16x16x32_f16(
                false, wbuf[cur][c].v, false, fa1[c].v, (short)0, acc1,
                false, false);
        }
        // acc[r] = A_full[e, 16t + r + 8*hi]; i = t>>1, j = 16*(t&1)+r+8*hi
        const int it = t >> 1;
        const int jb = (t & 1) * 8;       // index into hreg (j pre-mapped)
#pragma unroll
        for (int r = 0; r < 8; ++r) {
            P0[it] = fmaf(acc0[r], hreg0[jb + r], P0[it]);
            P1[it] = fmaf(acc1[r], hreg1[jb + r], P1[it]);
        }
    }

    // ---- b-path: m_b^T = W_b2^T x SB^T  (2 tiles x 4 K-chunks) ----
    // fb fragments built here (after the hot loop) to keep them out of the
    // main loop's live range.
    F16x16 fb0[4], fb1[4];
#pragma unroll
    for (int c = 0; c < 4; ++c) {
        const float* wq = w1Bp + (c * 32 + lane) * 16;
        const float* bq = b1Bp + (c * 32 + lane) * 16;
#pragma unroll
        for (int q = 0; q < 16; ++q) {
            float w = wq[q], b = bq[q];
            fb0[c].v[q] = (_Float16)selu_f(fmaf(s0, w, b));
            fb1[c].v[q] = (_Float16)selu_f(fmaf(s1, w, b));
        }
    }
#pragma unroll
    for (int T = 0; T < 2; ++T) {
        v8f acc0 = {}, acc1 = {};
#pragma unroll
        for (int c = 0; c < 4; ++c) {
            F16x16 wf;
            load_frag(wf, wb2p, T * 4 + c, lane);
            acc0 = __builtin_amdgcn_wmma_f32_16x16x32_f16(
                false, wf.v, false, fb0[c].v, (short)0, acc0, false, false);
            acc1 = __builtin_amdgcn_wmma_f32_16x16x32_f16(
                false, wf.v, false, fb1[c].v, (short)0, acc1, false, false);
        }
        if (hi) {
#pragma unroll
            for (int r = 0; r < 8; ++r) {
                P0[16 * T + 8 + r] += acc0[r];
                P1[16 * T + 8 + r] += acc1[r];
            }
        } else {
#pragma unroll
            for (int r = 0; r < 8; ++r) {
                P0[16 * T + r] += acc0[r];
                P1[16 * T + r] += acc1[r];
            }
        }
    }

    // ---- b_A2 . h term: b_A2[32x32] x h^T  (2 tiles, K=32) ----
    F16x16 fh0, fh1;
#pragma unroll
    for (int q = 0; q < 16; ++q) {
        fh0.v[q] = (_Float16)hreg0[q];
        fh1.v[q] = (_Float16)hreg1[q];
    }
#pragma unroll
    for (int T = 0; T < 2; ++T) {
        F16x16 wf;
        load_frag(wf, ba2p, T, lane);
        v8f acc0 = {}, acc1 = {};
        acc0 = __builtin_amdgcn_wmma_f32_16x16x32_f16(
            false, wf.v, false, fh0.v, (short)0, acc0, false, false);
        acc1 = __builtin_amdgcn_wmma_f32_16x16x32_f16(
            false, wf.v, false, fh1.v, (short)0, acc1, false, false);
        if (hi) {
#pragma unroll
            for (int r = 0; r < 8; ++r) {
                P0[16 * T + 8 + r] += acc0[r];
                P1[16 * T + 8 + r] += acc1[r];
            }
        } else {
#pragma unroll
            for (int r = 0; r < 8; ++r) {
                P0[16 * T + r] += acc0[r];
                P1[16 * T + r] += acc1[r];
            }
        }
    }

    // ---- combine half-lanes (xor-16), add b_b2, store 64B/lane/group ----
    const float* bb = b_b2 + hi * 16;
#pragma unroll
    for (int g = 0; g < 2; ++g) {
        const float* P = g ? P1 : P0;
        float orow[16];
#pragma unroll
        for (int u = 0; u < 16; ++u) {
            float lo = P[u]      + __shfl_xor(P[u],      16, 32);
            float hv = P[16 + u] + __shfl_xor(P[16 + u], 16, 32);
            orow[u] = (hi ? hv : lo) + bb[u];
        }
        float* op = out + (g ? e1 : e0) * 32 + hi * 16;
#pragma unroll
        for (int u = 0; u < 4; ++u) {
            float4 v;
            v.x = orow[4 * u + 0]; v.y = orow[4 * u + 1];
            v.z = orow[4 * u + 2]; v.w = orow[4 * u + 3];
            *(float4*)(op + 4 * u) = v;
        }
    }
}

extern "C" void kernel_launch(void* const* d_in, const int* in_sizes, int n_in,
                              void* d_out, int out_size, void* d_ws,
                              size_t ws_size, hipStream_t stream) {
    const float* h_i  = (const float*)d_in[0];
    const float* e_ij = (const float*)d_in[1];
    const float* W_A1 = (const float*)d_in[2];
    const float* b_A1 = (const float*)d_in[3];
    const float* W_A2 = (const float*)d_in[4];
    const float* b_A2 = (const float*)d_in[5];
    const float* W_b1 = (const float*)d_in[6];
    const float* b_b1 = (const float*)d_in[7];
    const float* W_b2 = (const float*)d_in[8];
    const float* b_b2 = (const float*)d_in[9];
    float* out = (float*)d_out;

    char* ws = (char*)d_ws;
    _Float16* wtA  = (_Float16*)(ws + OFF_WT);
    _Float16* wb2p = (_Float16*)(ws + OFF_WB2);
    _Float16* ba2p = (_Float16*)(ws + OFF_BA2);
    float* w1Ap = (float*)(ws + OFF_W1A);
    float* b1Ap = (float*)(ws + OFF_B1A);
    float* w1Bp = (float*)(ws + OFF_W1B);
    float* b1Bp = (float*)(ws + OFF_B1B);

    // Prologue: pack weights into WMMA fragment layout (runs every call).
    pack_wtA<<<512, 256, 0, stream>>>(W_A2, wtA);
    pack_aux<<<28, 256, 0, stream>>>(W_b2, b_A2, W_A1, b_A1, W_b1, b_b1,
                                     wb2p, ba2p, w1Ap, b1Ap, w1Bp, b1Bp);

    const int E = in_sizes[1];              // 262144
    const int waves_needed = E / 32;        // one wave per 32 edges (2 groups)
    const int blocks = waves_needed / 8;    // 8 waves (256 threads) per block
    edge_mlp_wmma<<<blocks, 256, 0, stream>>>(h_i, e_ij, wtA, wb2p, ba2p,
                                              w1Ap, b1Ap, w1Bp, b1Bp, b_b2,
                                              out);
}
